// AttentionLayer_1520418423069
// MI455X (gfx1250) — matrix operand order repaired and checked by reading, not compiled, not run on hardware
//
#include <hip/hip_runtime.h>

// ---------------------------------------------------------------------------
// Types for WMMA fragments (CDNA5 / gfx1250, wave32)
// ---------------------------------------------------------------------------
typedef __attribute__((ext_vector_type(16))) __bf16 v16bf;
typedef __attribute__((ext_vector_type(8)))  __bf16 v8bf;
typedef __attribute__((ext_vector_type(4)))  __bf16 v4bf;
typedef __attribute__((ext_vector_type(8)))  float  v8f;

// fp32 -> bf16 round-to-nearest-even, pure integer ops
__device__ __forceinline__ __bf16 f2bf(float f) {
  union { float f; unsigned u; } c; c.f = f;
  unsigned r = c.u + 0x7FFFu + ((c.u >> 16) & 1u);
  union { unsigned short s; __bf16 b; } o;
  o.s = (unsigned short)(r >> 16);
  return o.b;
}

// ---------------------------------------------------------------------------
// Straight fp32 -> bf16 conversion (4 elements / thread)
// ---------------------------------------------------------------------------
__global__ __launch_bounds__(256)
void cvt_f32_bf16(const float* __restrict__ in, __bf16* __restrict__ out, int n4) {
  int i = blockIdx.x * blockDim.x + threadIdx.x;
  if (i < n4) {
    float4 f = reinterpret_cast<const float4*>(in)[i];
    v4bf o;
    o[0] = f2bf(f.x); o[1] = f2bf(f.y); o[2] = f2bf(f.z); o[3] = f2bf(f.w);
    reinterpret_cast<v4bf*>(out)[i] = o;
  }
}

// ---------------------------------------------------------------------------
// Transposing fp32 -> bf16 conversion via LDS tile:
//   in : [R, C] f32 (batch stride sIn),  out: [C, R] bf16 (batch stride sOut)
// block (32,8), grid (C/32, R/32, batch); R,C multiples of 32 here.
// ---------------------------------------------------------------------------
__global__ __launch_bounds__(256)
void transpose_f32_bf16(const float* __restrict__ in, __bf16* __restrict__ out,
                        int R, int C, long long sIn, long long sOut) {
  __shared__ __bf16 tile[32][33];   // +1 pad: avoid LDS bank conflicts
  const float* ib = in  + (size_t)blockIdx.z * sIn;
  __bf16*      ob = out + (size_t)blockIdx.z * sOut;
  const int r0 = blockIdx.y * 32, c0 = blockIdx.x * 32;
  #pragma unroll
  for (int i = 0; i < 4; ++i) {
    int r = threadIdx.y + i * 8;
    tile[r][threadIdx.x] = f2bf(ib[(size_t)(r0 + r) * C + (c0 + threadIdx.x)]);
  }
  __syncthreads();
  #pragma unroll
  for (int i = 0; i < 4; ++i) {
    int c = threadIdx.y + i * 8;                 // transposed-out row (orig col)
    ob[(size_t)(c0 + c) * R + (r0 + threadIdx.x)] = tile[threadIdx.x][c];
  }
}

// ---------------------------------------------------------------------------
// bf16 WMMA GEMM, NT-only: C[b] = epilogue(A[b] @ Bt[b]^T (+bias))
//   A : [M,K] row-major bf16 (ld = K)
//   Bt: [N,K] row-major bf16 (ld = K)  — contiguous-K fragment loads
//   C : [M,N] (ld = N), bf16 or f32; SIG applies sigmoid.
// Block (32,4) = 4 waves; each wave owns a 64(M) x 32(N) tile:
// 8 accumulators, A fragments reused x2, B fragments reused x4.
// Addressing: lane-variant BYTE offsets are loop-invariant VGPRs; the
// uniform (SGPR) base pointers advance by 128 B per unrolled iteration, so
// loads lower to saddr+voffset(+imm) global_load_b128 with no per-lane
// address VALU in the loop. First K-step uses inline-0 C operand, so no
// accumulator zero-init block. K-loop is ping/pong double-buffered.
// Requires M % 256 == 0, N % 32 == 0, K % 64 == 0 (true for all stages).
// ---------------------------------------------------------------------------
template<bool SIG, bool OUT_BF16>
__global__ __launch_bounds__(128)
void wmma_gemm_nt(const __bf16* __restrict__ A, const __bf16* __restrict__ Bt,
                  const float* __restrict__ bias, void* __restrict__ Cv,
                  int N, int K, long long sA, long long sB, long long sC)
{
  const int lane = threadIdx.x;      // 0..31 (wave32)
  const int half = lane >> 4;
  const int l16  = lane & 15;
  const int n0   = blockIdx.x * 32;
  const int m0   = (blockIdx.y * 4 + threadIdx.y) * 64;
  const int bz   = blockIdx.z;

  // uniform base pointers (stay in SGPRs, advanced scalar-side)
  const char* pA = (const char*)(A  + (size_t)bz * sA);
  const char* pB = (const char*)(Bt + (size_t)bz * sB);

  // ---- loop-invariant lane-variant BYTE offsets ----
  // A 16x32 bf16 layout: lanes 0-15 row m hold K {0..15},
  // lanes 16-31 same row hold K {16..31}  (elements -> x2 bytes).
  const unsigned aLoB = half ? 32u : 0u;
  const unsigned aHiB = half ? 48u : 16u;
  unsigned aOffLo[4], aOffHi[4];
  #pragma unroll
  for (int mt = 0; mt < 4; ++mt) {
    unsigned rowB = (unsigned)(m0 + mt * 16 + l16) * (unsigned)K * 2u;
    aOffLo[mt] = rowB + aLoB;
    aOffHi[mt] = rowB + aHiB;
  }
  // B 32x16 layout: lane = column, half selects K 0..15 / 16..31 (contiguous).
  unsigned bOff[2];
  bOff[0] = ((unsigned)(n0 + l16) * (unsigned)K + 16u * half) * 2u;
  bOff[1] = bOff[0] + 32u * (unsigned)K;      // +16 rows * K elems * 2 B

  v8f acc[4][2];

  auto loadAk = [&](const char* p, v16bf (&a)[4]) {
    #pragma unroll
    for (int mt = 0; mt < 4; ++mt) {
      v8bf lo = *reinterpret_cast<const v8bf*>(p + aOffLo[mt]);
      v8bf hi = *reinterpret_cast<const v8bf*>(p + aOffHi[mt]);
      #pragma unroll
      for (int e = 0; e < 8; ++e) { a[mt][e] = lo[e]; a[mt][e + 8] = hi[e]; }
    }
  };
  auto loadBk = [&](const char* p, v16bf (&b)[2]) {
    b[0] = *reinterpret_cast<const v16bf*>(p + bOff[0]);
    b[1] = *reinterpret_cast<const v16bf*>(p + bOff[1]);
  };
  auto mma = [&](v16bf (&a)[4], v16bf (&b)[2]) {
    #pragma unroll
    for (int mt = 0; mt < 4; ++mt) {
      acc[mt][0] = __builtin_amdgcn_wmma_f32_16x16x32_bf16(
          false, a[mt], false, b[0], (short)0, acc[mt][0], false, false);
      acc[mt][1] = __builtin_amdgcn_wmma_f32_16x16x32_bf16(
          false, a[mt], false, b[1], (short)0, acc[mt][1], false, false);
    }
  };

  // ping/pong fragment sets (disjoint registers, never copied)
  v16bf a0[4], a1[4], b0[2], b1[2];

  // prologue: first two K-steps in flight; first MMA burst uses inline C=0
  loadAk(pA, a0);      loadBk(pB, b0);
  loadAk(pA + 64, a1); loadBk(pB + 64, b1);
  #pragma unroll
  for (int mt = 0; mt < 4; ++mt) {
    acc[mt][0] = __builtin_amdgcn_wmma_f32_16x16x32_bf16(
        false, a0[mt], false, b0[0], (short)0, (v8f){}, false, false);
    acc[mt][1] = __builtin_amdgcn_wmma_f32_16x16x32_bf16(
        false, a0[mt], false, b0[1], (short)0, (v8f){}, false, false);
  }

  const int steps = K / 64;                       // double-steps; K % 64 == 0
  for (int it = 0; it < steps - 1; ++it) {
    pA += 128; pB += 128;                         // uniform: scalar adds
    loadAk(pA, a0);      loadBk(pB, b0);
    mma(a1, b1);
    loadAk(pA + 64, a1); loadBk(pB + 64, b1);
    mma(a0, b0);
  }
  mma(a1, b1);                                    // last K-step

  // epilogue; C/D layout: VGPR r -> row 8*half + r, col = l16 (+16*nt)
  const float bi0 = bias ? bias[n0 + l16] : 0.0f;
  const float bi1 = bias ? bias[n0 + 16 + l16] : 0.0f;
  #pragma unroll
  for (int nt = 0; nt < 2; ++nt) {
    const float bi = nt ? bi1 : bi0;
    const int col = n0 + nt * 16 + l16;
    #pragma unroll
    for (int mt = 0; mt < 4; ++mt) {
      #pragma unroll
      for (int r = 0; r < 8; ++r) {
        int row = m0 + mt * 16 + half * 8 + r;
        float v = acc[mt][nt][r] + bi;
        if (SIG) v = 1.0f / (1.0f + __expf(-v));
        size_t idx = (size_t)bz * sC + (size_t)row * N + col;
        if (OUT_BF16) reinterpret_cast<__bf16*>(Cv)[idx] = f2bf(v);
        else          reinterpret_cast<float*>(Cv)[idx]  = v;
      }
    }
  }
}

// ---------------------------------------------------------------------------
// Host-side orchestration
// ---------------------------------------------------------------------------
extern "C" void kernel_launch(void* const* d_in, const int* in_sizes, int n_in,
                              void* d_out, int out_size, void* d_ws, size_t ws_size,
                              hipStream_t stream) {
  (void)in_sizes; (void)n_in; (void)out_size; (void)ws_size;

  constexpr int  Bn = 4, S = 2048, D = 512, H = 512;
  constexpr long long BS = (long long)Bn * S;      // 8192

  const float* x  = (const float*)d_in[0];
  const float* Wq = (const float*)d_in[1];
  const float* bq = (const float*)d_in[2];
  const float* Wk = (const float*)d_in[3];
  const float* bk = (const float*)d_in[4];
  const float* Wv = (const float*)d_in[5];
  const float* bv = (const float*)d_in[6];
  const float* Wd = (const float*)d_in[7];
  const float* bd = (const float*)d_in[8];

  // workspace carve-up
  char* ws = (char*)d_ws;
  size_t off = 0;
  auto carve = [&](size_t elems) {
    void* p = ws + off;
    off += (elems * sizeof(__bf16) + 255) & ~(size_t)255;
    return (__bf16*)p;
  };
  __bf16* xb   = carve((size_t)BS * D);      // x, bf16                     8 MB
  __bf16* xbT  = carve((size_t)BS * D);      // per-batch x^T [D,S]         8 MB
  __bf16* wqT  = carve((size_t)D * H);       // Wq^T [H,D]
  __bf16* wkT  = carve((size_t)D * H);
  __bf16* wvT  = carve((size_t)D * H);
  __bf16* wdT  = carve((size_t)S * H);       // Wd^T [H,S]                  2 MB
  __bf16* qb   = carve((size_t)BS * H);      //                             8 MB
  __bf16* kb   = carve((size_t)BS * H);
  __bf16* vb   = carve((size_t)BS * H);
  __bf16* qkb  = carve((size_t)Bn * S * S);  //                            32 MB
  __bf16* db   = carve((size_t)BS * H);
  __bf16* vwb  = carve((size_t)Bn * S * S);  //                            32 MB

  // ---- conversions / transposes
  {
    int n4 = (int)((size_t)BS * D / 4);
    cvt_f32_bf16<<<(n4 + 255) / 256, 256, 0, stream>>>(x, xb, n4);
  }
  const dim3 tb(32, 8);
  // x[b] [S,D] -> xbT[b] [D,S]
  transpose_f32_bf16<<<dim3(D / 32, S / 32, Bn), tb, 0, stream>>>(
      x, xbT, S, D, (long long)S * D, (long long)D * S);
  // W* [D,H] -> [H,D]
  transpose_f32_bf16<<<dim3(H / 32, D / 32, 1), tb, 0, stream>>>(
      Wq, wqT, D, H, 0, 0);
  transpose_f32_bf16<<<dim3(H / 32, D / 32, 1), tb, 0, stream>>>(
      Wk, wkT, D, H, 0, 0);
  transpose_f32_bf16<<<dim3(H / 32, D / 32, 1), tb, 0, stream>>>(
      Wv, wvT, D, H, 0, 0);
  // Wd [S,H] -> [H,S]
  transpose_f32_bf16<<<dim3(H / 32, S / 32, 1), tb, 0, stream>>>(
      Wd, wdT, S, H, 0, 0);

  const dim3 blk(32, 4);
  const long long SH = (long long)S * H, SS = (long long)S * S, SD = (long long)S * D;

  // Stage 1: q/k/v = x @ W + b   (M=8192, N=512, K=512)
  {
    dim3 grid(H / 32, (int)(BS / 256), 1);
    wmma_gemm_nt<false, true><<<grid, blk, 0, stream>>>(xb, wqT, bq, qb, H, D, 0, 0, 0);
    wmma_gemm_nt<false, true><<<grid, blk, 0, stream>>>(xb, wkT, bk, kb, H, D, 0, 0, 0);
    wmma_gemm_nt<false, true><<<grid, blk, 0, stream>>>(xb, wvT, bv, vb, H, D, 0, 0, 0);
  }
  // Stage 2: qk[b] = q[b] @ k[b]^T  (M=N=2048, K=512)
  wmma_gemm_nt<false, true><<<dim3(S / 32, S / 256, Bn), blk, 0, stream>>>(
      qb, kb, nullptr, qkb, S, H, SH, SH, SS);
  // Stage 3: dense[b] = sigmoid(qk[b] @ Wd + bd)  (M=2048, N=512, K=2048)
  wmma_gemm_nt<true, true><<<dim3(H / 32, S / 256, Bn), blk, 0, stream>>>(
      qkb, wdT, bd, db, H, S, SS, 0, SH);
  // Stage 4: vw[b] = dense[b] @ v[b]^T  (M=N=2048, K=512)
  wmma_gemm_nt<false, true><<<dim3(S / 32, S / 256, Bn), blk, 0, stream>>>(
      db, vb, nullptr, vwb, S, H, SH, SH, SS);
  // Stage 5: out[b] = vw[b] @ x[b]  (M=2048, N=512, K=2048, f32 out)
  wmma_gemm_nt<false, false><<<dim3(D / 32, S / 256, Bn), blk, 0, stream>>>(
      vwb, xbT, nullptr, d_out, D, S, SS, SD, SD);
}
